// GPT_82557861363991
// MI455X (gfx1250) — compile-verified, matrix operand order checked by reading, not run on hardware
//
#include <hip/hip_runtime.h>

// ---------------------------------------------------------------------------
// GPT-2-style forward pass for gfx1250 (MI455X).
// f16 WMMA (fp32 accumulate) + Tensor Data Mover staging of fp32 tiles.
// L=6, D=1024, H=16, HD=64, V=32000, B=2, S=1024, FF=4096, M = B*S = 2048.
// ---------------------------------------------------------------------------

typedef __attribute__((ext_vector_type(16))) _Float16     v16h;
typedef __attribute__((ext_vector_type(2)))  _Float16     v2h;
typedef __attribute__((ext_vector_type(8)))  float        v8f;
typedef __attribute__((ext_vector_type(4)))  unsigned int v4u;
typedef __attribute__((ext_vector_type(8)))  int          v8i;
typedef __attribute__((ext_vector_type(4)))  int          v4i;

union FragH { v16h v; unsigned int u[8]; };
union FragF { v8f  v; float        f[8]; };

__device__ __forceinline__ unsigned pkh(float a, float b) {
  return __builtin_bit_cast(unsigned, __builtin_amdgcn_cvt_pkrtz(a, b));
}

#if __has_builtin(__builtin_amdgcn_tensor_load_to_lds)
#define HAVE_TDM 1
#else
#define HAVE_TDM 0
#endif

#if __has_builtin(__builtin_amdgcn_s_wait_tensorcnt)
#define WAIT_TENSORCNT0() __builtin_amdgcn_s_wait_tensorcnt(0)
#else
#define WAIT_TENSORCNT0() asm volatile("s_wait_tensorcnt 0x0" ::: "memory")
#endif

// LDS byte offset of a __shared__ object (generic -> addrspace(3) -> int).
#define LDS_OFF(p) ((unsigned)(unsigned long long)(__attribute__((address_space(3))) const void*)(p))

#if HAVE_TDM
// One TDM descriptor: DMA a 2-D fp32 tile [tile1 rows][tile0 contiguous]
// from global (row stride `stride0` elements) into LDS contiguously.
// D# per CDNA5 ISA 8.3/8.4: count=1, type=2(image), data_size=2(4B),
// tensor_dim = tile_dim (tiles are always fully in-bounds here).
// This toolchain's builtin takes 6 args: (g0, g1, g2, g3, g4, cpol).
__device__ __forceinline__ void tdm_load_2d(unsigned lds_off, const float* gptr,
                                            unsigned tile0, unsigned tile1,
                                            unsigned long long stride0) {
  const unsigned long long ga = (unsigned long long)gptr;
  v4u g0;
  g0[0] = 1u;                                               // count=1
  g0[1] = lds_off;                                          // lds_addr (bytes)
  g0[2] = (unsigned)ga;                                     // global_addr[31:0]
  g0[3] = ((unsigned)(ga >> 32) & 0x01FFFFFFu) | (2u << 30);// addr[56:32] | type=2
  v8i g1;
  g1[0] = (int)(2u << 16);                                  // data_size = 4 bytes
  g1[1] = (int)((tile0 & 0xFFFFu) << 16);                   // tensor_dim0 lo16
  g1[2] = (int)(((tile0 >> 16) & 0xFFFFu) | ((tile1 & 0xFFFFu) << 16)); // dim0 hi | dim1 lo
  g1[3] = (int)(((tile1 >> 16) & 0xFFFFu) | ((tile0 & 0xFFFFu) << 16)); // dim1 hi | tile_dim0
  g1[4] = (int)(tile1 & 0xFFFFu);                           // tile_dim1, tile_dim2=0
  g1[5] = (int)(unsigned)(stride0 & 0xFFFFFFFFull);         // dim0_stride lo32
  g1[6] = (int)(unsigned)((stride0 >> 32) & 0xFFFFull);     // dim0_stride hi16
  g1[7] = 0;
  const v4i z4 = {0, 0, 0, 0};
  const v8i z8 = {0, 0, 0, 0, 0, 0, 0, 0};
  __builtin_amdgcn_tensor_load_to_lds(g0, g1, z4, z4, z8, 0);
}
#endif

// A-operand fragment (16x32 f16) from an fp32 LDS tile [row][ldk], contraction
// contiguous. CDNA5 7.12.2: lanes 0-15 row=lane, K {0..7,16..23};
// lanes 16-31 row=lane-16, K {8..15,24..31}. 4x ds_load_b128 + 8x cvt_pk.
__device__ __forceinline__ void frag_rows_f32(FragH& fr, const float* lds,
                                              int row0, int k0, int ldk, int lane) {
  const float* p = lds + (row0 + (lane & 15)) * ldk + k0 + ((lane & 16) ? 8 : 0);
  const float4 c0 = *(const float4*)(p);
  const float4 c1 = *(const float4*)(p + 4);
  const float4 c2 = *(const float4*)(p + 16);
  const float4 c3 = *(const float4*)(p + 20);
  fr.u[0] = pkh(c0.x, c0.y); fr.u[1] = pkh(c0.z, c0.w);
  fr.u[2] = pkh(c1.x, c1.y); fr.u[3] = pkh(c1.z, c1.w);
  fr.u[4] = pkh(c2.x, c2.y); fr.u[5] = pkh(c2.z, c2.w);
  fr.u[6] = pkh(c3.x, c3.y); fr.u[7] = pkh(c3.z, c3.w);
}

// B-operand fragment (32x16 f16) from an fp32 LDS tile [k][ldn] (contraction is
// the slow axis). Fixed column per lane -> dword addr = k*ldn + n, bank = n:
// conflict-free strided ds_load_b32 reads.
__device__ __forceinline__ void frag_cols_f32(FragH& fr, const float* lds,
                                              int col0, int ldn, int lane) {
  const float* p = lds + ((lane & 16) ? 8 : 0) * ldn + col0 + (lane & 15);
#pragma unroll
  for (int i = 0; i < 4; ++i) {
    fr.u[i]     = pkh(p[(2 * i) * ldn],      p[(2 * i + 1) * ldn]);
    fr.u[i + 4] = pkh(p[(16 + 2 * i) * ldn], p[(17 + 2 * i) * ldn]);
  }
}

// A-operand fragment from an f16 LDS tile (used for the softmax P round-trip).
__device__ __forceinline__ void frag_rows_f16(FragH& fr, const _Float16* lds,
                                              int row0, int ldk, int lane) {
  const _Float16* p = lds + (row0 + (lane & 15)) * ldk + ((lane & 16) ? 8 : 0);
#pragma unroll
  for (int i = 0; i < 4; ++i) {
    fr.u[i]     = *reinterpret_cast<const unsigned int*>(p + 2 * i);
    fr.u[i + 4] = *reinterpret_cast<const unsigned int*>(p + 16 + 2 * i);
  }
}

// ---------------------------------------------------------------------------
// Embedding: h[token] = tok_emb[x[token]] + pos_emb[token % S]
// ---------------------------------------------------------------------------
__global__ __launch_bounds__(256) void gpt_embed_kernel(
    const int* __restrict__ x, const float* __restrict__ tok,
    const float* __restrict__ pos, float* __restrict__ h) {
  const int t  = blockIdx.x;
  const int s  = t & 1023;
  const int id = x[t];
#pragma unroll
  for (int i = 0; i < 4; ++i) {
    const int c = threadIdx.x + (i << 8);
    h[(size_t)t * 1024 + c] = tok[(size_t)id * 1024 + c] + pos[(size_t)s * 1024 + c];
  }
}

// ---------------------------------------------------------------------------
// LayerNorm over D=1024 per token. grid = tokens, block = 256 (8 waves).
// ---------------------------------------------------------------------------
__global__ __launch_bounds__(256) void gpt_ln_kernel(
    const float* __restrict__ xin, const float* __restrict__ g,
    const float* __restrict__ be, float* __restrict__ y) {
  __shared__ float sred[16];
  const int t   = blockIdx.x;
  const int tid = threadIdx.x;
  const float* xr = xin + (size_t)t * 1024;
  float v[4];
  float s = 0.f, s2 = 0.f;
#pragma unroll
  for (int i = 0; i < 4; ++i) {
    v[i] = xr[tid + (i << 8)];
    s += v[i];
    s2 += v[i] * v[i];
  }
#pragma unroll
  for (int d = 1; d < 32; d <<= 1) {
    s  += __shfl_xor(s, d, 32);
    s2 += __shfl_xor(s2, d, 32);
  }
  const int wave = tid >> 5, lane = tid & 31;
  if (lane == 0) { sred[wave] = s; sred[8 + wave] = s2; }
  __syncthreads();
  if (tid < 32) {
    float a = (lane < 8) ? sred[lane] : 0.f;
    float b = (lane < 8) ? sred[8 + lane] : 0.f;
#pragma unroll
    for (int d = 1; d < 8; d <<= 1) {
      a += __shfl_xor(a, d, 32);
      b += __shfl_xor(b, d, 32);
    }
    if (lane == 0) { sred[0] = a; sred[1] = b; }
  }
  __syncthreads();
  const float mu  = sred[0] * (1.0f / 1024.0f);
  const float var = sred[1] * (1.0f / 1024.0f) - mu * mu;
  const float inv = rsqrtf(var + 1e-5f);
#pragma unroll
  for (int i = 0; i < 4; ++i) {
    const int c = tid + (i << 8);
    y[(size_t)t * 1024 + c] = (v[i] - mu) * inv * g[c] + be[c];
  }
}

// ---------------------------------------------------------------------------
// GEMM: C[M,N] = act(A[M,K] @ W[K,N] + bias) (+ resid).  ACT: 0 none, 1 GELU.
// 128 threads (4 waves), 64x64 block tile, wave tile 32x32 (2x2 WMMA), K-step
// 32.  Tiles staged fp32 in LDS by the TDM (one tensor_load_to_lds per tile),
// converted to f16 during fragment build.
// ---------------------------------------------------------------------------
template <int ACT, int RES>
__global__ __launch_bounds__(128) void gpt_gemm_kernel(
    const float* __restrict__ A, const float* __restrict__ W,
    const float* __restrict__ bias, const float* __restrict__ resid,
    float* __restrict__ C, int M, int N, int K) {
  __shared__ float sAf[64 * 32];  // [m][k] k-contiguous (TDM natural)
  __shared__ float sBf[32 * 64];  // [k][n] n-contiguous (TDM natural)
  const int tid  = threadIdx.x;
  const int lane = tid & 31;
  const int wave = tid >> 5;
  const int wm   = (wave >> 1) << 5;
  const int wn   = (wave & 1) << 5;
  const int bm   = blockIdx.y << 6;
  const int bn   = blockIdx.x << 6;

  FragF acc[2][2];
#pragma unroll
  for (int i = 0; i < 2; ++i)
#pragma unroll
    for (int j = 0; j < 2; ++j)
#pragma unroll
      for (int r = 0; r < 8; ++r) acc[i][j].f[r] = 0.f;

  for (int k0 = 0; k0 < K; k0 += 32) {
#if HAVE_TDM
    if (wave == 0) {
      tdm_load_2d(LDS_OFF(sAf), A + (size_t)bm * K + k0, 32, 64, (unsigned long long)K);
      tdm_load_2d(LDS_OFF(sBf), W + (size_t)k0 * N + bn, 64, 32, (unsigned long long)N);
      WAIT_TENSORCNT0();
    }
#else
#pragma unroll
    for (int i = 0; i < 16; ++i) {
      const int idx = tid + (i << 7);  // 0..2047
      sAf[idx] = A[(size_t)(bm + (idx >> 5)) * K + (k0 + (idx & 31))];
      sBf[idx] = W[(size_t)(k0 + (idx >> 6)) * N + (bn + (idx & 63))];
    }
#endif
    __syncthreads();
    FragH a[2], b[2];
    frag_rows_f32(a[0], sAf, wm,      0, 32, lane);
    frag_rows_f32(a[1], sAf, wm + 16, 0, 32, lane);
    frag_cols_f32(b[0], sBf, wn,      64, lane);
    frag_cols_f32(b[1], sBf, wn + 16, 64, lane);
#pragma unroll
    for (int i = 0; i < 2; ++i)
#pragma unroll
      for (int j = 0; j < 2; ++j)
        acc[i][j].v = __builtin_amdgcn_wmma_f32_16x16x32_f16(
            false, a[i].v, false, b[j].v, (short)0, acc[i][j].v, false, false);
    __syncthreads();
  }

  const int rowoff = (lane & 16) ? 8 : 0;
  const int coloff = lane & 15;
#pragma unroll
  for (int i = 0; i < 2; ++i) {
#pragma unroll
    for (int j = 0; j < 2; ++j) {
      const int col = bn + wn + (j << 4) + coloff;
      const float bv = bias ? bias[col] : 0.f;
#pragma unroll
      for (int r = 0; r < 8; ++r) {
        const int row = bm + wm + (i << 4) + rowoff + r;
        float val = acc[i][j].f[r] + bv;
        if (ACT == 1) val = 0.5f * val * (1.0f + erff(val * 0.70710678118f));
        if (RES) val += resid[(size_t)row * N + col];
        C[(size_t)row * N + col] = val;
      }
    }
  }
}

// ---------------------------------------------------------------------------
// Causal flash attention, one wave per (batch, head, 16-row q tile).
// q/k/v/o layout [B,S,D], D = H*64.  k-tiles of 32, online softmax.
// Q/K/V tiles staged fp32 in LDS by the TDM; scores & P@V on WMMA.
// grid = (S/16, H, B), block = 32.
// ---------------------------------------------------------------------------
__global__ __launch_bounds__(32) void gpt_attn_kernel(
    const float* __restrict__ q, const float* __restrict__ k,
    const float* __restrict__ v, float* __restrict__ o) {
  __shared__ float    sQf[16 * 64];  // [qrow][d]
  __shared__ float    sKf[32 * 64];  // [krow][d] (scores B operand: rows)
  __shared__ float    sVf[32 * 64];  // [krow][d] (P@V B operand: columns)
  __shared__ _Float16 sP[16 * 32];   // [qrow][krow] (P@V A operand)

  const int lane = threadIdx.x;
  const int qt = blockIdx.x, hh = blockIdx.y, bb = blockIdx.z;
  const size_t base = (size_t)bb * 1024 * 1024 + (size_t)hh * 64;
  const int q0 = qt << 4;

#if HAVE_TDM
  tdm_load_2d(LDS_OFF(sQf), q + base + (size_t)q0 * 1024, 64, 16, 1024ull);
  WAIT_TENSORCNT0();
#else
#pragma unroll
  for (int i = 0; i < 32; ++i) {
    const int idx = lane + (i << 5);
    sQf[idx] = q[base + (size_t)(q0 + (idx >> 6)) * 1024 + (idx & 63)];
  }
#endif
  __syncthreads();
  FragH aq0, aq1;
  frag_rows_f32(aq0, sQf, 0, 0,  64, lane);
  frag_rows_f32(aq1, sQf, 0, 32, 64, lane);

  FragF accO[4];
#pragma unroll
  for (int t = 0; t < 4; ++t)
#pragma unroll
    for (int r = 0; r < 8; ++r) accO[t].f[r] = 0.f;

  const int rowoff = (lane & 16) ? 8 : 0;
  const int coloff = lane & 15;
  float m8[8], l8[8];
#pragma unroll
  for (int r = 0; r < 8; ++r) { m8[r] = -1e30f; l8[r] = 0.f; }

  const int qmax = q0 + 15;
  for (int k0 = 0; k0 <= qmax; k0 += 32) {
#if HAVE_TDM
    tdm_load_2d(LDS_OFF(sKf), k + base + (size_t)k0 * 1024, 64, 32, 1024ull);
    tdm_load_2d(LDS_OFF(sVf), v + base + (size_t)k0 * 1024, 64, 32, 1024ull);
    WAIT_TENSORCNT0();
#else
#pragma unroll
    for (int i = 0; i < 64; ++i) {
      const int idx = lane + (i << 5);
      sKf[idx] = k[base + (size_t)(k0 + (idx >> 6)) * 1024 + (idx & 63)];
      sVf[idx] = v[base + (size_t)(k0 + (idx >> 6)) * 1024 + (idx & 63)];
    }
#endif
    __syncthreads();

    // Scores S = Q @ K^T : two 16x16 C tiles (k-cols 0..15, 16..31).
    FragF s0, s1;
#pragma unroll
    for (int r = 0; r < 8; ++r) { s0.f[r] = 0.f; s1.f[r] = 0.f; }
    FragH bk_;
    frag_rows_f32(bk_, sKf, 0, 0, 64, lane);
    s0.v = __builtin_amdgcn_wmma_f32_16x16x32_f16(false, aq0.v, false, bk_.v, (short)0, s0.v, false, false);
    frag_rows_f32(bk_, sKf, 0, 32, 64, lane);
    s0.v = __builtin_amdgcn_wmma_f32_16x16x32_f16(false, aq1.v, false, bk_.v, (short)0, s0.v, false, false);
    frag_rows_f32(bk_, sKf, 16, 0, 64, lane);
    s1.v = __builtin_amdgcn_wmma_f32_16x16x32_f16(false, aq0.v, false, bk_.v, (short)0, s1.v, false, false);
    frag_rows_f32(bk_, sKf, 16, 32, 64, lane);
    s1.v = __builtin_amdgcn_wmma_f32_16x16x32_f16(false, aq1.v, false, bk_.v, (short)0, s1.v, false, false);

    // Scale + causal mask + online softmax (16-lane butterflies for row stats).
    float tmax[8];
#pragma unroll
    for (int r = 0; r < 8; ++r) {
      const int qrow = q0 + r + rowoff;
      float x0 = s0.f[r] * 0.125f;
      float x1 = s1.f[r] * 0.125f;
      if (k0 + coloff > qrow) x0 = -1e30f;
      if (k0 + 16 + coloff > qrow) x1 = -1e30f;
      s0.f[r] = x0; s1.f[r] = x1;
      tmax[r] = fmaxf(x0, x1);
    }
#pragma unroll
    for (int d = 1; d < 16; d <<= 1)
#pragma unroll
      for (int r = 0; r < 8; ++r) tmax[r] = fmaxf(tmax[r], __shfl_xor(tmax[r], d, 32));

    float alpha[8], psum[8];
#pragma unroll
    for (int r = 0; r < 8; ++r) {
      const float newm = fmaxf(m8[r], tmax[r]);
      alpha[r] = __expf(m8[r] - newm);
      m8[r] = newm;
      const float p0 = __expf(s0.f[r] - newm);
      const float p1 = __expf(s1.f[r] - newm);
      s0.f[r] = p0; s1.f[r] = p1;
      psum[r] = p0 + p1;
    }
#pragma unroll
    for (int d = 1; d < 16; d <<= 1)
#pragma unroll
      for (int r = 0; r < 8; ++r) psum[r] += __shfl_xor(psum[r], d, 32);
#pragma unroll
    for (int r = 0; r < 8; ++r) l8[r] = l8[r] * alpha[r] + psum[r];
#pragma unroll
    for (int t = 0; t < 4; ++t)
#pragma unroll
      for (int r = 0; r < 8; ++r) accO[t].f[r] *= alpha[r];

    // P through LDS (f16) to become an A fragment, then accO += P @ V.
    __syncthreads();
#pragma unroll
    for (int r = 0; r < 8; ++r) {
      sP[((r + rowoff) << 5) + coloff]      = (_Float16)s0.f[r];
      sP[((r + rowoff) << 5) + 16 + coloff] = (_Float16)s1.f[r];
    }
    __syncthreads();
    FragH ap, bv_;
    frag_rows_f16(ap, sP, 0, 32, lane);
#pragma unroll
    for (int t = 0; t < 4; ++t) {
      frag_cols_f32(bv_, sVf, t << 4, 64, lane);
      accO[t].v = __builtin_amdgcn_wmma_f32_16x16x32_f16(
          false, ap.v, false, bv_.v, (short)0, accO[t].v, false, false);
    }
    __syncthreads();
  }

  // Normalize and write O.
#pragma unroll
  for (int t = 0; t < 4; ++t)
#pragma unroll
    for (int r = 0; r < 8; ++r) {
      const int row = q0 + r + rowoff;
      const int d = (t << 4) + coloff;
      o[base + (size_t)row * 1024 + d] = accO[t].f[r] / l8[r];
    }
}

// ---------------------------------------------------------------------------
// Host-side orchestration.
// ---------------------------------------------------------------------------
extern "C" void kernel_launch(void* const* d_in, const int* in_sizes, int n_in,
                              void* d_out, int out_size, void* d_ws, size_t ws_size,
                              hipStream_t stream) {
  (void)in_sizes; (void)n_in; (void)out_size; (void)ws_size;
  constexpr int Lc = 6, Dc = 1024, Vc = 32000, Bc = 2, Sc = 1024, Fc = 4096;
  constexpr int Mc = Bc * Sc;

  const int*   x    = (const int*)d_in[0];
  const float* tok  = (const float*)d_in[1];
  const float* pos  = (const float*)d_in[2];
  const float* Wq   = (const float*)d_in[3];
  const float* bq   = (const float*)d_in[4];
  const float* Wk   = (const float*)d_in[5];
  const float* bk   = (const float*)d_in[6];
  const float* Wv   = (const float*)d_in[7];
  const float* bv   = (const float*)d_in[8];
  const float* Wo   = (const float*)d_in[9];
  const float* bo   = (const float*)d_in[10];
  const float* W1   = (const float*)d_in[11];
  const float* b1   = (const float*)d_in[12];
  const float* W2   = (const float*)d_in[13];
  const float* b2   = (const float*)d_in[14];
  const float* ln1g = (const float*)d_in[15];
  const float* ln1b = (const float*)d_in[16];
  const float* ln2g = (const float*)d_in[17];
  const float* ln2b = (const float*)d_in[18];
  const float* lnfg = (const float*)d_in[19];
  const float* lnfb = (const float*)d_in[20];
  const float* Wout = (const float*)d_in[21];
  float* out = (float*)d_out;

  float* ws  = (float*)d_ws;
  const size_t td = (size_t)Mc * Dc;
  float* h   = ws;
  float* tmp = h + td;
  float* qb  = tmp + td;
  float* kb  = qb + td;
  float* vb  = kb + td;
  float* ob  = vb + td;
  float* ff1 = ob + td;

  const dim3 blk(128);
  const dim3 gD(Dc / 64, Mc / 64);
  const dim3 gF(Fc / 64, Mc / 64);
  const dim3 gV(Vc / 64, Mc / 64);
  const dim3 gAttn(Sc / 16, 16, Bc);

  gpt_embed_kernel<<<Mc, 256, 0, stream>>>(x, tok, pos, h);

  for (int l = 0; l < Lc; ++l) {
    const size_t wdd = (size_t)l * Dc * Dc;
    const size_t wdf = (size_t)l * Dc * Fc;
    gpt_gemm_kernel<0, 0><<<gD, blk, 0, stream>>>(h, Wq + wdd, bq + l * Dc, nullptr, qb, Mc, Dc, Dc);
    gpt_gemm_kernel<0, 0><<<gD, blk, 0, stream>>>(h, Wk + wdd, bk + l * Dc, nullptr, kb, Mc, Dc, Dc);
    gpt_gemm_kernel<0, 0><<<gD, blk, 0, stream>>>(h, Wv + wdd, bv + l * Dc, nullptr, vb, Mc, Dc, Dc);
    gpt_attn_kernel<<<gAttn, 32, 0, stream>>>(qb, kb, vb, ob);
    gpt_gemm_kernel<0, 1><<<gD, blk, 0, stream>>>(ob, Wo + wdd, bo + l * Dc, h, tmp, Mc, Dc, Dc);
    gpt_ln_kernel<<<Mc, 256, 0, stream>>>(tmp, ln1g + l * Dc, ln1b + l * Dc, h);
    gpt_gemm_kernel<1, 0><<<gF, blk, 0, stream>>>(h, W1 + wdf, b1 + l * Fc, nullptr, ff1, Mc, Fc, Dc);
    gpt_gemm_kernel<0, 1><<<gD, blk, 0, stream>>>(ff1, W2 + wdf, b2 + l * Dc, h, tmp, Mc, Dc, Fc);
    gpt_ln_kernel<<<Mc, 256, 0, stream>>>(tmp, ln2g + l * Dc, ln2b + l * Dc, h);
  }

  gpt_ln_kernel<<<Mc, 256, 0, stream>>>(h, lnfg, lnfb, tmp);
  gpt_gemm_kernel<0, 0><<<gV, blk, 0, stream>>>(tmp, Wout, nullptr, nullptr, out, Mc, Vc, Dc);
}
